// ClassicalSelfAttention_65481071401153
// MI455X (gfx1250) — matrix operand order repaired and checked by reading, not compiled, not run on hardware
//
#include <hip/hip_runtime.h>
#include <hip/hip_bf16.h>
#include <math.h>

// Self-attention: B=1, N=4096, D=768, H=12, Hd=64, fp32 in/out.
// bf16 operands + fp32-accumulating v_wmma_f32_16x16x32_bf16, flash attention,
// double-buffered LDS tiles filled with GLOBAL_LOAD_ASYNC_TO_LDS_B128
// (ASYNCcnt-tracked, no VGPR round-trip), s_wait_asynccnt + barrier sync.

#define TOKENS  4096
#define DMODEL  768
#define NHEADS  12
#define HDIM    64
#define QKVOUT  2304

typedef __bf16 bf16;
typedef __attribute__((ext_vector_type(16))) __bf16 v16bf;
typedef __attribute__((ext_vector_type(8)))  __bf16 v8bf;
typedef __attribute__((ext_vector_type(8)))  float  v8f;

__device__ __forceinline__ v16bf cat8(v8bf lo, v8bf hi) {
  return __builtin_shufflevector(lo, hi, 0,1,2,3,4,5,6,7,8,9,10,11,12,13,14,15);
}

__device__ __forceinline__ v8f wmma_bf16(v16bf a, v16bf b, v8f c) {
  // D(f32 16x16) = A(16x32 bf16) * B(32x16 bf16) + C
  return __builtin_amdgcn_wmma_f32_16x16x32_bf16(false, a, false, b, (short)0, c,
                                                 false, false);
}

// Generic LDS pointer -> wave-relative LDS byte address (flat-LDS addresses
// carry the LDS offset in bits [31:0] per the CDNA5 aperture mapping).
__device__ __forceinline__ unsigned lds_addr(const void* p) {
  return (unsigned)(size_t)p;
}

// Async 16B global -> LDS copy, per-lane addresses, tracked by ASYNCcnt.
__device__ __forceinline__ void async_ld16(unsigned lds_off, const bf16* gp) {
  asm volatile("global_load_async_to_lds_b128 %0, %1, off"
               :: "v"(lds_off), "v"(gp)
               : "memory");
}

__device__ __forceinline__ void wait_async0() {
  asm volatile("s_wait_asynccnt 0x0" ::: "memory");
}

// ---------------------------------------------------------------- fp32->bf16
__global__ void cvt_f32_bf16(const float* __restrict__ in, bf16* __restrict__ out,
                             int n) {
  int i = blockIdx.x * blockDim.x + threadIdx.x;
  int stride = gridDim.x * blockDim.x;
  for (; i < n; i += stride) out[i] = (bf16)in[i];
}

// ---------------------------------------------------------------- GEMM
// C[M x Ncols] = A[M x K] * B[Ncols x K]^T, bf16 operands, fp32 accum.
// Workgroup tile 128x64, 8 waves (each: 16 rows x 64 cols), K-step 32,
// double-buffered async LDS staging.
// MODE 0: QKV epilogue (scatter bf16 into per-head Q, K, transposed V)
// MODE 1: out-projection epilogue (add bias, write fp32)
template <int MODE>
__global__ __launch_bounds__(256)
void gemm_bf16_k(const bf16* __restrict__ A, const bf16* __restrict__ B, int K,
                 bf16* __restrict__ qb, bf16* __restrict__ kb,
                 bf16* __restrict__ vtb,
                 const float* __restrict__ bias, float* __restrict__ outf) {
  constexpr int LDA = 40;  // 32 + 8 bf16 pad (16B) to spread LDS banks
  constexpr int LDB = 40;
  __shared__ bf16 sA[2][128 * LDA];
  __shared__ bf16 sB[2][64 * LDB];

  const int tid  = threadIdx.x;
  const int wave = tid >> 5;
  const int lane = tid & 31;
  const int l16  = lane & 15;
  const int lhi  = lane >> 4;
  const int rowBase = blockIdx.x * 128;
  const int colBase = blockIdx.y * 64;

  const v8f zv = {0.f,0.f,0.f,0.f,0.f,0.f,0.f,0.f};
  v8f acc[4];
#pragma unroll
  for (int i = 0; i < 4; ++i) acc[i] = zv;

  // Issue async staging of one K-slice into LDS buffer `buf`.
  auto stage = [&](int buf, int kt) {
    // A tile (128x32): 512 16B chunks, 2 per thread.
    for (int c = tid; c < 512; c += 256) {
      int r = c >> 2, ko = (c & 3) << 3;
      async_ld16(lds_addr(&sA[buf][r * LDA + ko]),
                 &A[(size_t)(rowBase + r) * K + kt + ko]);
    }
    // B tile (64x32): 256 chunks, 1 per thread.
    int r = tid >> 2, ko = (tid & 3) << 3;
    async_ld16(lds_addr(&sB[buf][r * LDB + ko]),
               &B[(size_t)(colBase + r) * K + kt + ko]);
  };

  const int nk = K / 32;
  stage(0, 0);
  for (int it = 0; it < nk; ++it) {
    const int cur = it & 1;
    wait_async0();       // own async loads (<= current buffer) have landed
    __syncthreads();     // everyone's loads landed; old buffer fully consumed
    if (it + 1 < nk) stage(cur ^ 1, (it + 1) * 32);  // overlap with compute

    // A fragment: lane holds row m, K chunks at lhi*8 and lhi*8+16.
    const int am = wave * 16 + l16;
    v16bf afrag = cat8(*(v8bf*)&sA[cur][am * LDA + lhi * 8],
                       *(v8bf*)&sA[cur][am * LDA + lhi * 8 + 16]);
#pragma unroll
    for (int nb = 0; nb < 4; ++nb) {
      // B fragment: lane holds col n, K chunks at lhi*8 and lhi*8+16.
      const int bn = nb * 16 + l16;
      v16bf bfrag = cat8(*(v8bf*)&sB[cur][bn * LDB + lhi * 8],
                         *(v8bf*)&sB[cur][bn * LDB + lhi * 8 + 16]);
      acc[nb] = wmma_bf16(afrag, bfrag, acc[nb]);
    }
    // No end barrier: next iteration's top barrier separates this compute
    // from the next overwrite of the other buffer.
  }

  // Epilogue. C layout: vgpr r of lane -> (m = r + 8*lhi, n = l16).
#pragma unroll
  for (int nb = 0; nb < 4; ++nb) {
#pragma unroll
    for (int r = 0; r < 8; ++r) {
      const int row = rowBase + wave * 16 + r + 8 * lhi;
      const int col = colBase + nb * 16 + l16;
      const float v = acc[nb][r];
      if (MODE == 0) {
        const int which = col / DMODEL;      // 0=Q 1=K 2=V
        const int d = col % DMODEL;
        const int h = d >> 6, hd = d & 63;
        if (which == 0)
          qb[((size_t)h * TOKENS + row) * HDIM + hd] = (bf16)v;
        else if (which == 1)
          kb[((size_t)h * TOKENS + row) * HDIM + hd] = (bf16)v;
        else  // V stored transposed per head: [h][hd][token]
          vtb[((size_t)h * HDIM + hd) * TOKENS + row] = (bf16)v;
      } else {
        outf[(size_t)row * DMODEL + col] = v + bias[col];
      }
    }
  }
}

// ---------------------------------------------------------------- flash attn
// One head, 64 query rows per workgroup (4 waves x 16 rows).
// Key blocks of 64, double-buffered async K/Vt staging; S via WMMA, online
// softmax fp32, P through LDS, O accumulated fp32 via WMMA.
__global__ __launch_bounds__(128)
void flash_attn_k(const bf16* __restrict__ qb, const bf16* __restrict__ kb,
                  const bf16* __restrict__ vtb, bf16* __restrict__ ob) {
  constexpr int LDK = 72;  // 64 + 8 pad
  __shared__ bf16 sK[2][64 * LDK];  // [key][hd]
  __shared__ bf16 sV[2][64 * LDK];  // [hd][key]  (V transposed)
  __shared__ bf16 sP[64 * LDK];     // [wave*16+m][key]

  const int tid  = threadIdx.x;
  const int wave = tid >> 5;
  const int lane = tid & 31;
  const int l16  = lane & 15;
  const int lhi  = lane >> 4;
  const int h     = blockIdx.y;
  const int qbase = blockIdx.x * 64;

  // Q fragments straight from global (per-lane 16B loads, registered once).
  const bf16* qrow = qb + ((size_t)h * TOKENS + qbase + wave * 16 + l16) * HDIM;
  v16bf qf[2];
#pragma unroll
  for (int s = 0; s < 2; ++s)
    qf[s] = cat8(*(const v8bf*)&qrow[s * 32 + lhi * 8],
                 *(const v8bf*)&qrow[s * 32 + lhi * 8 + 16]);

  const v8f zv = {0.f,0.f,0.f,0.f,0.f,0.f,0.f,0.f};
  v8f oacc[4];
  float mstat[8], lstat[8];
#pragma unroll
  for (int i = 0; i < 4; ++i) oacc[i] = zv;
#pragma unroll
  for (int r = 0; r < 8; ++r) { mstat[r] = -3.0e38f; lstat[r] = 0.f; }

  const float scale = 0.125f;  // 1/sqrt(64)

  // Async staging of one key block into buffer `buf`.
  auto stage = [&](int buf, int kk) {
    for (int c = tid; c < 256; c += 128) {
      int r = c >> 2, ko = (c & 3) << 3;
      async_ld16(lds_addr(&sK[buf][r * LDK + ko]),
                 &kb[((size_t)h * TOKENS + kk + r) * HDIM + ko]);
      async_ld16(lds_addr(&sV[buf][r * LDK + ko]),
                 &vtb[((size_t)h * HDIM + r) * TOKENS + kk + ko]);
    }
  };

  stage(0, 0);
  for (int kk = 0; kk < TOKENS; kk += 64) {
    const int cur = (kk >> 6) & 1;
    wait_async0();
    __syncthreads();
    if (kk + 64 < TOKENS) stage(cur ^ 1, kk + 64);  // overlap with compute

    // S = Q K^T  (16 q-rows x 64 keys per wave; Hd=64 -> 2 k-steps)
    v8f sacc[4];
#pragma unroll
    for (int i = 0; i < 4; ++i) sacc[i] = zv;
#pragma unroll
    for (int nb = 0; nb < 4; ++nb) {
      const int key = nb * 16 + l16;
#pragma unroll
      for (int s = 0; s < 2; ++s) {
        v16bf kf = cat8(*(v8bf*)&sK[cur][key * LDK + s * 32 + lhi * 8],
                        *(v8bf*)&sK[cur][key * LDK + s * 32 + lhi * 8 + 16]);
        sacc[nb] = wmma_bf16(qf[s], kf, sacc[nb]);
      }
    }

    // Online softmax. Row r lives in vgpr r across the 16 lanes of a
    // half-wave -> reduce with shfl_xor 1,2,4,8.
    float alpha[8];
#pragma unroll
    for (int r = 0; r < 8; ++r) {
      float rmax = -3.0e38f;
#pragma unroll
      for (int nb = 0; nb < 4; ++nb) {
        float sv = sacc[nb][r] * scale;
        sacc[nb][r] = sv;
        rmax = fmaxf(rmax, sv);
      }
      rmax = fmaxf(rmax, __shfl_xor(rmax, 1, 32));
      rmax = fmaxf(rmax, __shfl_xor(rmax, 2, 32));
      rmax = fmaxf(rmax, __shfl_xor(rmax, 4, 32));
      rmax = fmaxf(rmax, __shfl_xor(rmax, 8, 32));
      const float mnew = fmaxf(mstat[r], rmax);
      alpha[r] = __expf(mstat[r] - mnew);
      mstat[r] = mnew;

      float rsum = 0.f;
#pragma unroll
      for (int nb = 0; nb < 4; ++nb) {
        float p = __expf(sacc[nb][r] - mnew);
        sacc[nb][r] = p;
        rsum += p;
      }
      rsum += __shfl_xor(rsum, 1, 32);
      rsum += __shfl_xor(rsum, 2, 32);
      rsum += __shfl_xor(rsum, 4, 32);
      rsum += __shfl_xor(rsum, 8, 32);
      lstat[r] = lstat[r] * alpha[r] + rsum;
#pragma unroll
      for (int nb = 0; nb < 4; ++nb) oacc[nb][r] *= alpha[r];
    }

    // P (C layout) -> LDS (per-wave private region; LDS ops are in-order).
#pragma unroll
    for (int nb = 0; nb < 4; ++nb)
#pragma unroll
      for (int r = 0; r < 8; ++r)
        sP[(wave * 16 + r + 8 * lhi) * LDK + nb * 16 + l16] = (bf16)sacc[nb][r];

    // O += P V  (K-dim = 64 keys -> 2 k-steps)
    const int pm = wave * 16 + l16;
    v16bf pf[2];
#pragma unroll
    for (int s = 0; s < 2; ++s)
      pf[s] = cat8(*(v8bf*)&sP[pm * LDK + s * 32 + lhi * 8],
                   *(v8bf*)&sP[pm * LDK + s * 32 + lhi * 8 + 16]);
#pragma unroll
    for (int nb = 0; nb < 4; ++nb) {
      const int hd = nb * 16 + l16;
#pragma unroll
      for (int s = 0; s < 2; ++s) {
        v16bf vf = cat8(*(v8bf*)&sV[cur][hd * LDK + s * 32 + lhi * 8],
                        *(v8bf*)&sV[cur][hd * LDK + s * 32 + lhi * 8 + 16]);
        oacc[nb] = wmma_bf16(pf[s], vf, oacc[nb]);
      }
    }
    // Next iteration's top barrier separates this compute from the next
    // overwrite of the other buffer.
  }

  // Finalize: O /= l, write bf16 [token][D] ready for the out-projection.
#pragma unroll
  for (int r = 0; r < 8; ++r) {
    const float inv = 1.0f / lstat[r];
    const int row = qbase + wave * 16 + r + 8 * lhi;
#pragma unroll
    for (int nb = 0; nb < 4; ++nb) {
      const int col = h * HDIM + nb * 16 + l16;
      ob[(size_t)row * DMODEL + col] = (bf16)(oacc[nb][r] * inv);
    }
  }
}

// ---------------------------------------------------------------- launcher
extern "C" void kernel_launch(void* const* d_in, const int* in_sizes, int n_in,
                              void* d_out, int out_size, void* d_ws,
                              size_t ws_size, hipStream_t stream) {
  const float* x     = (const float*)d_in[0];
  const float* w_qkv = (const float*)d_in[1];
  const float* w_out = (const float*)d_in[2];
  const float* b_out = (const float*)d_in[3];
  float* out = (float*)d_out;

  // Workspace carve-out (~36 MB total).
  char* p = (char*)d_ws;
  bf16* xb    = (bf16*)p; p += (size_t)TOKENS * DMODEL * 2;
  bf16* wqkvb = (bf16*)p; p += (size_t)QKVOUT * DMODEL * 2;
  bf16* woutb = (bf16*)p; p += (size_t)DMODEL * DMODEL * 2;
  bf16* qb    = (bf16*)p; p += (size_t)NHEADS * TOKENS * HDIM * 2;
  bf16* kb    = (bf16*)p; p += (size_t)NHEADS * TOKENS * HDIM * 2;
  bf16* vtb   = (bf16*)p; p += (size_t)NHEADS * TOKENS * HDIM * 2;
  bf16* ob    = (bf16*)p; p += (size_t)TOKENS * DMODEL * 2;

  cvt_f32_bf16<<<2048, 256, 0, stream>>>(x, xb, TOKENS * DMODEL);
  cvt_f32_bf16<<<2048, 256, 0, stream>>>(w_qkv, wqkvb, QKVOUT * DMODEL);
  cvt_f32_bf16<<<1024, 256, 0, stream>>>(w_out, woutb, DMODEL * DMODEL);

  // QKV projection: M=4096, N=2304, K=768 -> grid 32 x 36.
  gemm_bf16_k<0><<<dim3(TOKENS / 128, QKVOUT / 64), 256, 0, stream>>>(
      xb, wqkvb, DMODEL, qb, kb, vtb, nullptr, nullptr);

  // Flash attention: 64 q-rows per WG x 12 heads.
  flash_attn_k<<<dim3(TOKENS / 64, NHEADS), 128, 0, stream>>>(qb, kb, vtb, ob);

  // Output projection + bias: M=4096, N=768, K=768 -> grid 32 x 12.
  gemm_bf16_k<1><<<dim3(TOKENS / 128, DMODEL / 64), 256, 0, stream>>>(
      ob, woutb, DMODEL, nullptr, nullptr, nullptr, b_out, out);
}
